// GlobalModel_A_26302379720747
// MI455X (gfx1250) — matrix operand order repaired and checked by reading, not compiled, not run on hardware
//
#include <hip/hip_runtime.h>
#include <hip/hip_bf16.h>

// ---------------------------------------------------------------------------
// GlobalModel attention-pool for MI455X (gfx1250, wave32).
//
// Roofline: dominant traffic = e (819 MB) + x (51 MB) + indices (~7 MB)
// => ~38 us at 23.3 TB/s. The naive e@Wk GEMM (26 GFLOPs fp32) would be
// compute-bound, so we refactor the bilinear attention score:
//     score(e) = (e@Wk + bk) . q[g]  =  e . (Wk q[g]) + bk . q[g]
// Per-graph vectors w[g]=Wk q[g] ([128,128] total, WGP$/L2-resident) reduce
// the streaming pass to one length-128 dot per row -> pure bandwidth.
// Streamed rows use non-temporal loads (TH=NT) so the 870 MB stream does not
// evict the hot 64 KB w-tables from L2/WGP$.
// Small exact matrix stages (q = u@Wq, w = q@Wk^T, final concat@Wu) run on
// v_wmma_f32_16x16x32_f16 (f32 accumulate).
// ---------------------------------------------------------------------------

typedef __attribute__((ext_vector_type(16))) _Float16 v16h;
typedef __attribute__((ext_vector_type(8)))  float    v8f;
typedef __attribute__((ext_vector_type(4)))  float    v4f;  // clang vector:
                                                            // OK for
                                                            // nontemporal ld

__device__ __forceinline__ v8f wmma_f16(v16h a, v16h b, v8f c) {
  // (neg_a, A, neg_b, B, c_mod, C, reuse_a, reuse_b)
  return __builtin_amdgcn_wmma_f32_16x16x32_f16(false, a, false, b,
                                                (short)0, c, false, false);
}

// A-matrix 16x32 (MxK) f16 layout: lanes 0-15 row M=lane hold
// K=0..15 (elems 0..15); lanes 16-31 hold K=16..31 for the same rows
// (matches the B k order).
__device__ __forceinline__ v16h load_a_f32(const float* S, int ld, int m0,
                                           int k0, int lane) {
  const int m  = m0 + (lane & 15);
  const int hi = lane >> 4;
  v16h a;
#pragma unroll
  for (int j = 0; j < 16; ++j) {
    const int k = k0 + j + hi * 16;
    a[j] = (_Float16)S[m * ld + k];
  }
  return a;
}

// B-matrix 32x16 (KxN) f16 layout: N = lane&15; lanes 0-15 hold K=0..15
// (elems 0..15), lanes 16-31 hold K=16..31.  trans=true reads S[n][k].
__device__ __forceinline__ v16h load_b_f32(const float* S, int ld, bool trans,
                                           int k0, int n0, int lane) {
  const int n  = n0 + (lane & 15);
  const int hi = lane >> 4;
  v16h b;
#pragma unroll
  for (int j = 0; j < 16; ++j) {
    const int k = k0 + j + hi * 16;
    b[j] = (_Float16)(trans ? S[n * ld + k] : S[k * ld + n]);
  }
  return b;
}

// ---------------------------------------------------------------------------
// Kernel 0: zero the aggregation buffers (deterministic across replays).
// ---------------------------------------------------------------------------
__global__ void __launch_bounds__(256) zero_kernel(float* p, int n) {
  int i = blockIdx.x * blockDim.x + threadIdx.x;
  if (i < n) p[i] = 0.0f;
}

// ---------------------------------------------------------------------------
// Kernel 1: per-graph precompute (one block, 8 waves, WMMA).
//   q   = u @ Wq + bq            [128 x 64]
//   w   = q @ Wk^T               [128 x 128]   (w[g][f] = sum_h q[g][h] Wk[f][h])
//   c   = q . bk                 [128]
// for both edge- and node-attention parameter sets.
// ---------------------------------------------------------------------------
__global__ void __launch_bounds__(256) precompute_kernel(
    const float* __restrict__ u,
    const float* __restrict__ Wq_e, const float* __restrict__ bq_e,
    const float* __restrict__ Wk_e, const float* __restrict__ bk_e,
    const float* __restrict__ Wq_x, const float* __restrict__ bq_x,
    const float* __restrict__ Wk_x, const float* __restrict__ bk_x,
    float* __restrict__ w_e, float* __restrict__ c_e,
    float* __restrict__ w_x, float* __restrict__ c_x) {
  __shared__ float qsh[128 * 64];  // 32 KB
  const int lane = threadIdx.x & 31;
  const int wid  = threadIdx.x >> 5;  // 8 waves -> 8 M-tiles of 16
  const int n16  = lane & 15;
  const int hi   = lane >> 4;

  for (int side = 0; side < 2; ++side) {
    const float* Wq = side ? Wq_x : Wq_e;
    const float* bq = side ? bq_x : bq_e;
    const float* Wk = side ? Wk_x : Wk_e;
    const float* bk = side ? bk_x : bk_e;
    float* wout = side ? w_x : w_e;
    float* cout = side ? c_x : c_e;

    // q = u @ Wq + bq  : M=128 (tile=wid), N=64 (4 tiles), K=128 (4 steps)
    for (int nt = 0; nt < 4; ++nt) {
      v8f acc = {};
#pragma unroll
      for (int kt = 0; kt < 4; ++kt) {
        v16h a = load_a_f32(u, 128, wid * 16, kt * 32, lane);
        v16h b = load_b_f32(Wq, 64, false, kt * 32, nt * 16, lane);
        acc = wmma_f16(a, b, acc);
      }
#pragma unroll
      for (int r = 0; r < 8; ++r) {  // C/D: VGPR r = row r (+8 for hi lanes)
        const int m = wid * 16 + r + 8 * hi;
        const int n = nt * 16 + n16;
        qsh[m * 64 + n] = acc[r] + bq[n];
      }
    }
    __syncthreads();

    // w = q @ Wk^T : M=128, N=128 (8 tiles), K=64 (2 steps)
    for (int nt = 0; nt < 8; ++nt) {
      v8f acc = {};
#pragma unroll
      for (int kt = 0; kt < 2; ++kt) {
        v16h a = load_a_f32(qsh, 64, wid * 16, kt * 32, lane);
        v16h b = load_b_f32(Wk, 64, true, kt * 32, nt * 16, lane);  // B=Wk^T
        acc = wmma_f16(a, b, acc);
      }
#pragma unroll
      for (int r = 0; r < 8; ++r) {
        const int m = wid * 16 + r + 8 * hi;
        wout[m * 128 + nt * 16 + n16] = acc[r];
      }
    }

    // c[g] = q[g,:] . bk
    if (threadIdx.x < 128) {
      float s = 0.0f;
      for (int h = 0; h < 64; ++h) s += qsh[threadIdx.x * 64 + h] * bk[h];
      cout[threadIdx.x] = s;
    }
    __syncthreads();  // protect qsh before next side overwrites it
  }
}

// ---------------------------------------------------------------------------
// Kernels 2/3: streaming attention + segment-sum.
// Each wave owns 4 consecutive rows per iteration (2 KB contiguous stream,
// 4+ independent b128 loads in flight for MLP). 32 lanes x v4f covers one
// 512 B row. Rows are loaded non-temporally (streamed once). Partial dots are
// wave-reduced with shuffles; a*row accumulates into a per-block [128,128]
// LDS accumulator (ds_add_f32), flushed once with global atomics
// (avoids ~200M global atomics).
// ---------------------------------------------------------------------------
__global__ void __launch_bounds__(256) attn_scatter_kernel(
    const float* __restrict__ feat,   // [M,128]
    const int*   __restrict__ src,    // [M] edge->node, or nullptr (identity)
    const int*   __restrict__ batch,  // [N] node->graph
    const float* __restrict__ w,      // [128,128] per-graph score vectors
    const float* __restrict__ cvec,   // [128] per-graph score constants
    float*       __restrict__ agg,    // [128,128] global accumulator
    int M) {
  __shared__ float acc[128 * 128];  // 64 KB -> 5 blocks / 320 KB WGP
  for (int i = threadIdx.x; i < 128 * 128; i += 256) acc[i] = 0.0f;
  __syncthreads();

  const int lane    = threadIdx.x & 31;
  const int wid     = threadIdx.x >> 5;
  const int wstride = gridDim.x * 8 * 4;  // rows per grid sweep

  for (int base = (blockIdx.x * 8 + wid) * 4; base < M; base += wstride) {
    const int rows = (M - base < 4) ? (M - base) : 4;

    // Phase 1: issue all streamed row loads + index chains (MLP).
    v4f v[4];
    int g[4];
#pragma unroll
    for (int r = 0; r < 4; ++r) {
      if (r < rows) {
        const int row  = base + r;
        const int node = src ? src[row] : row;      // uniform per wave
        g[r] = batch[node];                         // uniform -> broadcast
        v[r] = __builtin_nontemporal_load(
            (const v4f*)(feat + (size_t)row * 128) + lane);  // TH=NT
      }
    }
    // Phase 2: gather per-graph score vectors (hot in WGP$/L2).
    v4f wv[4];
#pragma unroll
    for (int r = 0; r < 4; ++r)
      if (r < rows) wv[r] = *((const v4f*)(w + g[r] * 128) + lane);

    // Phase 3: dot, wave-reduce, sigmoid, LDS scatter-accumulate.
#pragma unroll
    for (int r = 0; r < 4; ++r) {
      if (r < rows) {
        float p = v[r].x * wv[r].x + v[r].y * wv[r].y +
                  v[r].z * wv[r].z + v[r].w * wv[r].w;
#pragma unroll
        for (int off = 16; off > 0; off >>= 1) p += __shfl_xor(p, off, 32);
        const float a = 1.0f / (1.0f + __expf(-(p + cvec[g[r]])));
        float* dst = acc + g[r] * 128 + lane * 4;
        atomicAdd(dst + 0, a * v[r].x);   // ds_add_f32 (no return)
        atomicAdd(dst + 1, a * v[r].y);
        atomicAdd(dst + 2, a * v[r].z);
        atomicAdd(dst + 3, a * v[r].w);
      }
    }
  }
  __syncthreads();

  for (int i = threadIdx.x; i < 128 * 128; i += 256) {
    const float t = acc[i];
    if (t != 0.0f) atomicAdd(&agg[i], t);
  }
}

// ---------------------------------------------------------------------------
// Kernel 4: out = concat(x_agg, e_agg, u) @ Wu + bu   (128x384 @ 384x128)
// One block, 8 waves, 12 K-steps of 32 picking the A-source per chunk.
// ---------------------------------------------------------------------------
__global__ void __launch_bounds__(256) final_gemm_kernel(
    const float* __restrict__ x_agg, const float* __restrict__ e_agg,
    const float* __restrict__ u, const float* __restrict__ Wu,
    const float* __restrict__ bu, float* __restrict__ out) {
  const int lane = threadIdx.x & 31;
  const int wid  = threadIdx.x >> 5;
  const int n16  = lane & 15;
  const int hi   = lane >> 4;

  for (int nt = 0; nt < 8; ++nt) {
    v8f acc = {};
#pragma unroll
    for (int kt = 0; kt < 12; ++kt) {
      const int k0 = kt * 32;
      const float* srcA;
      int off;
      if (k0 < 128)      { srcA = x_agg; off = k0; }
      else if (k0 < 256) { srcA = e_agg; off = k0 - 128; }
      else               { srcA = u;     off = k0 - 256; }
      v16h a = load_a_f32(srcA, 128, wid * 16, off, lane);
      v16h b = load_b_f32(Wu, 128, false, k0, nt * 16, lane);
      acc = wmma_f16(a, b, acc);
    }
#pragma unroll
    for (int r = 0; r < 8; ++r) {
      const int m = wid * 16 + r + 8 * hi;
      const int n = nt * 16 + n16;
      out[m * 128 + n] = acc[r] + bu[n];
    }
  }
}

// ---------------------------------------------------------------------------
extern "C" void kernel_launch(void* const* d_in, const int* in_sizes, int n_in,
                              void* d_out, int out_size, void* d_ws,
                              size_t ws_size, hipStream_t stream) {
  const float* x    = (const float*)d_in[0];
  const float* e    = (const float*)d_in[1];
  const float* u    = (const float*)d_in[2];
  const int* edge_index = (const int*)d_in[3];  // [2,E]; row 0 = src
  const int* batch      = (const int*)d_in[4];  // [N]
  const float* Wk_e = (const float*)d_in[5];
  const float* bk_e = (const float*)d_in[6];
  const float* Wq_e = (const float*)d_in[7];
  const float* bq_e = (const float*)d_in[8];
  const float* Wk_x = (const float*)d_in[9];
  const float* bk_x = (const float*)d_in[10];
  const float* Wq_x = (const float*)d_in[11];
  const float* bq_x = (const float*)d_in[12];
  const float* Wu   = (const float*)d_in[13];
  const float* bu   = (const float*)d_in[14];
  float* out = (float*)d_out;

  const int N = in_sizes[0] / 128;
  const int E = in_sizes[1] / 128;

  // workspace layout (floats)
  float* ws    = (float*)d_ws;
  float* w_e   = ws;              // [128,128]
  float* w_x   = ws + 16384;      // [128,128]
  float* c_e   = ws + 32768;      // [128]
  float* c_x   = ws + 32896;      // [128]
  float* e_agg = ws + 33024;      // [128,128]
  float* x_agg = ws + 49408;      // [128,128]  (agg region contiguous: 32768)

  zero_kernel<<<(32768 + 255) / 256, 256, 0, stream>>>(e_agg, 32768);

  precompute_kernel<<<1, 256, 0, stream>>>(u, Wq_e, bq_e, Wk_e, bk_e,
                                           Wq_x, bq_x, Wk_x, bk_x,
                                           w_e, c_e, w_x, c_x);

  // edge pass: src = edge_index[0][:]  (first E entries)
  attn_scatter_kernel<<<512, 256, 0, stream>>>(e, edge_index, batch, w_e, c_e,
                                               e_agg, E);
  // node pass: identity indexing
  attn_scatter_kernel<<<256, 256, 0, stream>>>(x, nullptr, batch, w_x, c_x,
                                               x_agg, N);

  final_gemm_kernel<<<1, 256, 0, stream>>>(x_agg, e_agg, u, Wu, bu, out);
}
